// RGCN_VAE_82746839924849
// MI455X (gfx1250) — matrix operand order repaired and checked by reading, not compiled, not run on hardware
//
#include <hip/hip_runtime.h>
#include <hip/hip_bf16.h>
#include <stdint.h>
#include <stddef.h>

// Problem constants (match reference)
#define N_NODES 8192
#define E_EDGES 131072
#define R_REL   4
#define K1      384      // 6 * IN_EMBD
#define H1C     256      // H1
#define NCAT    1280     // 256 * (1 + R): [root | W[0..3]] concatenated
#define DLAT    128      // D

typedef __attribute__((ext_vector_type(16))) __bf16 v16bf;
typedef __attribute__((ext_vector_type(8)))  __bf16 v8bf;
typedef __attribute__((ext_vector_type(8)))  float  v8f;
typedef __attribute__((ext_vector_type(4)))  float  f32x4;

__device__ __forceinline__ unsigned short f2bf(float f) {
  union { float f; unsigned int u; } c; c.f = f;
  unsigned int u = c.u;
  unsigned int r = (u + 0x7FFFu + ((u >> 16) & 1u)) >> 16;  // RNE
  return (unsigned short)r;
}

__device__ __forceinline__ float sigmoidf(float x) {
  return 1.0f / (1.0f + __expf(-x));
}

__device__ __forceinline__ void atomAddF(float* p, float v) {
  __hip_atomic_fetch_add(p, v, __ATOMIC_RELAXED, __HIP_MEMORY_SCOPE_AGENT);
}

__device__ __forceinline__ unsigned int hash_u32(unsigned int x) {
  x ^= x >> 16; x *= 0x7feb352dU;
  x ^= x >> 15; x *= 0x846ca68bU;
  x ^= x >> 16; return x;
}

// gfx1250 async global->LDS copy (ASYNCcnt-tracked), per CDNA5 ISA.
// VDST = LDS byte address VGPR, VADDR = 64-bit global address.
__device__ __forceinline__ void async_b128_to_lds(unsigned lds_byte_addr, const void* gptr) {
  asm volatile("global_load_async_to_lds_b128 %0, %1, off"
               :: "v"(lds_byte_addr), "v"((unsigned long long)(size_t)gptr)
               : "memory");
}
__device__ __forceinline__ void wait_asynccnt0() {
  asm volatile("s_wait_asynccnt 0x0" ::: "memory");
}

// -------------------- fills --------------------
__global__ void fill_f32x4_nt(f32x4* __restrict__ p, float v, size_t n4) {
  size_t i = (size_t)blockIdx.x * blockDim.x + threadIdx.x;
  size_t stride = (size_t)gridDim.x * blockDim.x;
  f32x4 val = {v, v, v, v};
  for (; i < n4; i += stride) __builtin_nontemporal_store(val, p + i);
}

__global__ void fill_f32(float* __restrict__ p, float v, size_t n) {
  size_t i = (size_t)blockIdx.x * blockDim.x + threadIdx.x;
  size_t stride = (size_t)gridDim.x * blockDim.x;
  for (; i < n; i += stride) p[i] = v;
}

// -------------------- embedding gather -> bf16 X --------------------
__global__ __launch_bounds__(384)
void gather_embed(const int* __restrict__ x,
                  const float* __restrict__ e0, const float* __restrict__ e1,
                  const float* __restrict__ e2, const float* __restrict__ e3,
                  const float* __restrict__ e4, const float* __restrict__ e5,
                  unsigned short* __restrict__ Xbf) {
  int n = blockIdx.x;
  int t = threadIdx.x;         // 0..383
  int f = t >> 6;              // field 0..5
  int c = t & 63;
  const float* tab = (f == 0) ? e0 : (f == 1) ? e1 : (f == 2) ? e2
                   : (f == 3) ? e3 : (f == 4) ? e4 : e5;
  int idx = x[n * 6 + f];
  Xbf[(size_t)n * K1 + t] = f2bf(tab[idx * 64 + c]);
}

// -------------------- weight concat + transpose -> bf16 (NCAT x K) --------------------
__global__ void build_wt(const float* __restrict__ W,    // (R, K, 256)
                         const float* __restrict__ root, // (K, 256)
                         unsigned short* __restrict__ Wt, int K) {
  int i = blockIdx.x * blockDim.x + threadIdx.x;
  if (i >= NCAT * K) return;
  int nn = i / K, k = i - nn * K;
  float v;
  if (nn < 256) {
    v = root[(size_t)k * 256 + nn];
  } else {
    int r = (nn >> 8) - 1, col = nn & 255;
    v = W[((size_t)r * K + k) * 256 + col];
  }
  Wt[(size_t)nn * K + k] = f2bf(v);
}

__global__ void build_bias(const float* __restrict__ b, float* __restrict__ biascat) {
  int i = blockIdx.x * blockDim.x + threadIdx.x;
  if (i < NCAT) biascat[i] = (i < 256) ? b[i] : 0.0f;
}

// -------------------- per-(relation,dst) inverse counts --------------------
__global__ void count_edges(const int* __restrict__ ei, const int* __restrict__ et,
                            float* __restrict__ cnt) {
  int e = blockIdx.x * blockDim.x + threadIdx.x;
  if (e >= E_EDGES) return;
  int d = ei[E_EDGES + e];
  int r = et[e];
  atomAddF(&cnt[(size_t)r * N_NODES + d], 1.0f);
}

__global__ void inv_counts(float* __restrict__ cnt) {
  int i = blockIdx.x * blockDim.x + threadIdx.x;
  if (i < R_REL * N_NODES) cnt[i] = 1.0f / fmaxf(cnt[i], 1.0f);
}

// -------------------- WMMA NT GEMM: C[M x NC] = A[M x K] * B[NC x K]^T (+bias) ---------
// A, B bf16 row-major with K contiguous.  Block = 8 waves = one 16-row m-strip x 128
// output columns; one 16x16 tile per wave, K-loop step 32.
// The 16xK A strip (contiguous 32*K bytes, shared by all 8 waves) is staged into LDS
// once per block via GLOBAL_LOAD_ASYNC_TO_LDS_B128 + s_wait_asynccnt; WMMA A fragments
// are then served by ds_load_b128.  B fragments (distinct per wave) load from global.
// A frag (16x32): lane L -> row m0+(L&15), chunks [kb,kb+8) / [kb+16,kb+24), kb=(L>=16)*8.
// B frag (32x16): lane L -> col n0+(L&15), chunks [kb,kb+8) / [kb+16,kb+24), kb=(L>=16)*8.
template <bool SIG, bool NT>
__global__ __launch_bounds__(256)
void wmma_gemm_nt(const unsigned short* __restrict__ A,
                  const unsigned short* __restrict__ B,
                  const float* __restrict__ bias,
                  float* __restrict__ C,
                  int K, int NC) {
  __shared__ unsigned short Asmem[16 * K1];   // max K = 384 -> 12 KB
  int wave = threadIdx.x >> 5;
  int lane = threadIdx.x & 31;
  int ntile = blockIdx.x * 8 + wave;
  int m0 = blockIdx.y * 16;

  // cooperative async stage of the contiguous 16xK A strip
  {
    const char* src = (const char*)(A + (size_t)m0 * K);
    unsigned lds0 = (unsigned)(size_t)(&Asmem[0]);
    int chunks = 2 * K;                       // 16-byte chunks: 32*K bytes total
    for (int c = threadIdx.x; c < chunks; c += 256)
      async_b128_to_lds(lds0 + c * 16, src + c * 16);
    wait_asynccnt0();
  }
  __syncthreads();

  if (ntile * 16 < NC) {
    int n0 = ntile * 16;
    int r16 = lane & 15, hi = lane >> 4;
    const unsigned short* als  = &Asmem[r16 * K + hi * 8];
    const unsigned short* brow = B + (size_t)(n0 + r16) * K + hi * 8;

    v8f cacc = {};
    for (int k = 0; k < K; k += 32) {
      union { v16bf v; v8bf h[2]; } af, bf;
      af.h[0] = *reinterpret_cast<const v8bf*>(als + k);
      af.h[1] = *reinterpret_cast<const v8bf*>(als + k + 16);
      bf.h[0] = *reinterpret_cast<const v8bf*>(brow + k);
      bf.h[1] = *reinterpret_cast<const v8bf*>(brow + k + 16);
      if (k + 32 < K) __builtin_prefetch(brow + k + 32, 0, 3);  // WGP-scope prefetch
      cacc = __builtin_amdgcn_wmma_f32_16x16x32_bf16(
          false, af.v, false, bf.v, (short)0, cacc, false, false);
    }
    float bcol = bias ? bias[n0 + r16] : 0.0f;
    size_t cbase = (size_t)(m0 + hi * 8) * NC + (n0 + r16);
    #pragma unroll
    for (int i = 0; i < 8; ++i) {
      float v = cacc[i] + bcol;
      if (SIG) v = sigmoidf(v);
      if (NT) __builtin_nontemporal_store(v, C + cbase + (size_t)i * NC);
      else    C[cbase + (size_t)i * NC] = v;
    }
  }
}

// -------------------- edge message aggregation (one block per edge) --------------------
__global__ __launch_bounds__(256)
void edge_aggregate(const int* __restrict__ ei, const int* __restrict__ et,
                    const float* __restrict__ G,    // (N, NCAT): [hlin | h_r0..h_r3]
                    const float* __restrict__ invc, // (R, N)
                    float* __restrict__ acc) {      // (N, 256)
  int e = blockIdx.x;
  int t = threadIdx.x;
  int s = ei[e], d = ei[E_EDGES + e], r = et[e];
  float w = invc[(size_t)r * N_NODES + d];
  float v = G[(size_t)s * NCAT + 256 * (1 + r) + t] * w;
  atomAddF(&acc[(size_t)d * 256 + t], v);
}

// -------------------- combine + sigmoid -> bf16 hidden --------------------
__global__ __launch_bounds__(256)
void combine_sigmoid_bf16(const float* __restrict__ G, const float* __restrict__ acc,
                          unsigned short* __restrict__ Hbf) {
  int i = blockIdx.x * blockDim.x + threadIdx.x;
  if (i >= N_NODES * 256) return;
  int n = i >> 8, c = i & 255;
  Hbf[i] = f2bf(sigmoidf(G[(size_t)n * NCAT + c] + acc[i]));
}

// -------------------- layer-2 combine + VAE reparameterization --------------------
__global__ __launch_bounds__(256)
void vae_reparam(const float* __restrict__ G, const float* __restrict__ acc,
                 float* __restrict__ mu_out, float* __restrict__ lv_out,
                 unsigned short* __restrict__ Zbf) {
  int i = blockIdx.x * blockDim.x + threadIdx.x;
  if (i >= N_NODES * DLAT) return;
  int n = i >> 7, d = i & 127;
  float mu = sigmoidf(G[(size_t)n * NCAT + d]       + acc[(size_t)n * 256 + d]);
  float lv = sigmoidf(G[(size_t)n * NCAT + 128 + d] + acc[(size_t)n * 256 + 128 + d]);
  mu_out[i] = mu;
  lv_out[i] = lv;
  float stdv = __expf(0.5f * lv);
  unsigned int ha = hash_u32(2u * (unsigned)i + 0x9E3779B9u);
  unsigned int hb = hash_u32(2u * (unsigned)i + 1u);
  float u1 = (float)(ha >> 8) * (1.0f / 16777216.0f) + 1.0e-12f;
  float u2 = (float)(hb >> 8) * (1.0f / 16777216.0f);
  float eps = sqrtf(-2.0f * __logf(u1)) * __cosf(6.28318530718f * u2);
  Zbf[i] = f2bf(eps * stdv + mu);
}

// -------------------- adjacency scatter --------------------
__global__ void scatter_A(const int* __restrict__ ei, float* __restrict__ A) {
  int e = blockIdx.x * blockDim.x + threadIdx.x;
  if (e >= E_EDGES) return;
  int s = ei[e], d = ei[E_EDGES + e];
  A[(size_t)s * N_NODES + d] = 1.0f;
}

// -------------------- host launcher --------------------
extern "C" void kernel_launch(void* const* d_in, const int* in_sizes, int n_in,
                              void* d_out, int out_size, void* d_ws, size_t ws_size,
                              hipStream_t stream) {
  (void)in_sizes; (void)n_in; (void)out_size; (void)ws_size;
  const int*   x     = (const int*)d_in[0];
  const int*   ei    = (const int*)d_in[1];
  const int*   et    = (const int*)d_in[2];
  const float* e0    = (const float*)d_in[3];
  const float* e1    = (const float*)d_in[4];
  const float* e2    = (const float*)d_in[5];
  const float* e3    = (const float*)d_in[6];
  const float* e4    = (const float*)d_in[7];
  const float* e5    = (const float*)d_in[8];
  const float* W1    = (const float*)d_in[9];
  const float* root1 = (const float*)d_in[10];
  const float* b1    = (const float*)d_in[11];
  const float* W2    = (const float*)d_in[12];
  const float* root2 = (const float*)d_in[13];
  const float* b2    = (const float*)d_in[14];

  float* out  = (float*)d_out;
  float* A    = out;
  float* Ahat = out + (size_t)N_NODES * N_NODES;
  float* muo  = out + 2ull * N_NODES * N_NODES;
  float* lvo  = muo + (size_t)N_NODES * DLAT;

  // workspace carve-up (256B aligned)
  char* w = (char*)d_ws;
  auto carve = [&](size_t bytes) { char* p = w; w += (bytes + 255) & ~(size_t)255; return p; };
  float*          G     = (float*)carve(sizeof(float) * N_NODES * NCAT);      // 41.9 MB
  float*          accb  = (float*)carve(sizeof(float) * N_NODES * 256);       //  8.4 MB
  unsigned short* Hbf   = (unsigned short*)carve(2ull * N_NODES * K1);        //  6.3 MB (X / h1)
  unsigned short* Wt1   = (unsigned short*)carve(2ull * NCAT * K1);           //  1.0 MB
  unsigned short* Wt2   = (unsigned short*)carve(2ull * NCAT * H1C);          //  0.7 MB
  float*          bias1 = (float*)carve(sizeof(float) * NCAT);
  float*          bias2 = (float*)carve(sizeof(float) * NCAT);
  float*          cnt   = (float*)carve(sizeof(float) * R_REL * N_NODES);
  unsigned short* Zbf   = (unsigned short*)carve(2ull * N_NODES * DLAT);      //  2.1 MB

  // ---- static prep (recomputed every call; deterministic) ----
  gather_embed<<<N_NODES, 384, 0, stream>>>(x, e0, e1, e2, e3, e4, e5, Hbf);
  build_wt<<<(NCAT * K1 + 255) / 256, 256, 0, stream>>>(W1, root1, Wt1, K1);
  build_wt<<<(NCAT * H1C + 255) / 256, 256, 0, stream>>>(W2, root2, Wt2, H1C);
  build_bias<<<(NCAT + 255) / 256, 256, 0, stream>>>(b1, bias1);
  build_bias<<<(NCAT + 255) / 256, 256, 0, stream>>>(b2, bias2);
  fill_f32<<<128, 256, 0, stream>>>(cnt, 0.0f, (size_t)R_REL * N_NODES);
  count_edges<<<E_EDGES / 256, 256, 0, stream>>>(ei, et, cnt);
  inv_counts<<<(R_REL * N_NODES + 255) / 256, 256, 0, stream>>>(cnt);

  // ---- layer 1: G = X @ [root1|W1]^T + b1cat ----
  wmma_gemm_nt<false, false><<<dim3(NCAT / 128, N_NODES / 16), 256, 0, stream>>>(
      Hbf, Wt1, bias1, G, K1, NCAT);
  fill_f32x4_nt<<<2048, 256, 0, stream>>>((f32x4*)accb, 0.0f, (size_t)N_NODES * 256 / 4);
  edge_aggregate<<<E_EDGES, 256, 0, stream>>>(ei, et, G, cnt, accb);
  combine_sigmoid_bf16<<<N_NODES * 256 / 256, 256, 0, stream>>>(G, accb, Hbf);

  // ---- layer 2 ----
  wmma_gemm_nt<false, false><<<dim3(NCAT / 128, N_NODES / 16), 256, 0, stream>>>(
      Hbf, Wt2, bias2, G, H1C, NCAT);
  fill_f32x4_nt<<<2048, 256, 0, stream>>>((f32x4*)accb, 0.0f, (size_t)N_NODES * 256 / 4);
  edge_aggregate<<<E_EDGES, 256, 0, stream>>>(ei, et, G, cnt, accb);

  // ---- VAE heads: mu, logvar, Z ----
  vae_reparam<<<N_NODES * DLAT / 256, 256, 0, stream>>>(G, accb, muo, lvo, Zbf);

  // ---- A: zero (NT) + scatter ones ----
  fill_f32x4_nt<<<8192, 256, 0, stream>>>((f32x4*)A, 0.0f, (size_t)N_NODES * N_NODES / 4);
  scatter_A<<<E_EDGES / 256, 256, 0, stream>>>(ei, A);

  // ---- A_hat = sigmoid(Z Z^T): K=128, NT stores ----
  wmma_gemm_nt<true, true><<<dim3(N_NODES / 128, N_NODES / 16), 256, 0, stream>>>(
      Zbf, Zbf, nullptr, Ahat, DLAT, N_NODES);
}
